// MEGConv_18373870092602
// MI455X (gfx1250) — matrix operand order repaired and checked by reading, not compiled, not run on hardware
//
#include <hip/hip_runtime.h>
#include <hip/hip_bf16.h>

// ---------------------------------------------------------------------------
// MEGNet-style graph network on MI455X (gfx1250), wave32 + WMMA f16.
//   atom_new = MLP96([atom, mean_bond(atom), glob[atom_mol]])
//   bond_new = MLP128([bond, atom_new[b0], atom_new[b1], glob[bond_mol]])
//   glob_new = MLP96([glob, mean(atom_new|mol), mean(bond_new|mol)])
// All GEMMs via v_wmma_f32_16x16x32_f16 (f16 in, f32 accumulate).
// Softplus via bare v_exp_f32/v_log_f32 (no libm denorm guards).
// ---------------------------------------------------------------------------

typedef __attribute__((ext_vector_type(16))) _Float16 v16h_t;
typedef __attribute__((ext_vector_type(8)))  _Float16 v8h_t;
typedef __attribute__((ext_vector_type(8)))  float    v8f_t;

__device__ __forceinline__ void lds_fence() {
    asm volatile("s_wait_dscnt 0x0" ::: "memory");
}

// softplus(x) = max(x,0) + ln(1 + exp(-|x|)), with bare HW transcendentals:
// v_exp_f32 computes 2^x (fold log2e into the argument); v_log_f32 is log2
// (scale by ln2). Argument of log is in [1,2] -> no denorm guard needed.
__device__ __forceinline__ float softplus_f(float x) {
    const float LOG2E = 1.4426950408889634f;
    const float LN2   = 0.6931471805599453f;
    float e = __builtin_amdgcn_exp2f(-__builtin_fabsf(x) * LOG2E);
    float l = __builtin_amdgcn_logf(1.f + e) * LN2;
    return fmaxf(x, 0.f) + l;
}

__device__ __forceinline__ v8f_t wmma_f16(v16h_t a, v16h_t b, v8f_t c) {
    return __builtin_amdgcn_wmma_f32_16x16x32_f16(
        /*neg_a=*/false, a, /*neg_b=*/false, b,
        /*c_mod=*/(short)0, c, /*reuse_a=*/false, /*reuse_b=*/false);
}

// A-matrix fragment (16x32 f16 tile of a row-major LDS tile):
// lane holds row M=lane&15; lanes>=16 see K+8; halves 0..7 -> K=kb..kb+7,
// halves 8..15 -> K=kb+16..kb+23 (ISA 05_wmma 16-bit A layout).
__device__ __forceinline__ v16h_t load_a16(const _Float16* tile, int strideH,
                                           int kc, int lane) {
    int m  = lane & 15;
    int kb = (lane & 16) ? 8 : 0;
    const _Float16* p = tile + m * strideH + kc * 32 + kb;
    v8h_t lo = *(const v8h_t*)(p);
    v8h_t hi = *(const v8h_t*)(p + 16);
    v16h_t r;
#pragma unroll
    for (int i = 0; i < 8; ++i) { r[i] = lo[i]; r[i + 8] = hi[i]; }
    return r;
}

// Pre-packed B fragment: one contiguous 32-byte LDS load per lane.
__device__ __forceinline__ v16h_t wfrag(const _Float16* Wp, int kc, int ntiles,
                                        int nt, int lane) {
    return ((const v16h_t*)Wp)[(kc * ntiles + nt) * 32 + lane];
}

// Pack a row-major f32 weight [Kdim][Ndim] into fragment-ready f16 layout:
// B layout (matches A): elems 0..7 -> K=kb..kb+7, 8..15 -> K=kb+16..kb+23, kb=8 for lanes 16-31,
// N = lane&15 (+16*nt).
__device__ __forceinline__ void pack_weights(const float* __restrict__ W,
                                             _Float16* dst, int Kdim, int Ndim,
                                             int tid, int nthreads) {
    int ntiles = Ndim >> 4;
    int kch    = Kdim >> 5;
    int total  = kch * ntiles * 32 * 16;
    for (int i = tid; i < total; i += nthreads) {
        int j    = i & 15;
        int lane = (i >> 4) & 31;
        int t    = i >> 9;
        int nt   = t % ntiles;
        int kc   = t / ntiles;
        int r    = j >> 1, hi = j & 1;
        int K = kc * 32 + (j & 7) + ((j & 8) ? 16 : 0) + ((lane & 16) ? 8 : 0); (void)r; (void)hi;
        int N = nt * 16 + (lane & 15);
        dst[i] = (_Float16)W[(long)K * Ndim + N];
    }
}

// Three-layer MLP on one 16-row tile (per-wave private staging, no barriers).
// A: [16][KCH*32] f16, H: [16][64] f16 scratch, O: [16][32] f32 result.
template <int KCH>
__device__ __forceinline__ void mlp3(const _Float16* A,
                                     const _Float16* W1p, const float* b1,
                                     const _Float16* W2p, const float* b2,
                                     const _Float16* W3p, const float* b3,
                                     _Float16* H, float* O, int lane) {
    const int nlo  = lane & 15;
    const int mrow = (lane & 16) ? 8 : 0;

    v16h_t a[KCH];
#pragma unroll
    for (int kc = 0; kc < KCH; ++kc) a[kc] = load_a16(A, KCH * 32, kc, lane);

    // Layer 1: KIN -> 64, softplus
#pragma unroll
    for (int nt = 0; nt < 4; ++nt) {
        int   ncol = nt * 16 + nlo;
        float bb   = b1[ncol];
        v8f_t c;
#pragma unroll
        for (int r = 0; r < 8; ++r) c[r] = bb;   // accumulate bias in C
#pragma unroll
        for (int kc = 0; kc < KCH; ++kc)
            c = wmma_f16(a[kc], wfrag(W1p, kc, 4, nt, lane), c);
#pragma unroll
        for (int r = 0; r < 8; ++r)
            H[(mrow + r) * 64 + ncol] = (_Float16)softplus_f(c[r]);
    }
    lds_fence();

    // Layer 2: 64 -> 64, softplus (read fragments before overwriting H)
    v16h_t h0 = load_a16(H, 64, 0, lane);
    v16h_t h1 = load_a16(H, 64, 1, lane);
#pragma unroll
    for (int nt = 0; nt < 4; ++nt) {
        int   ncol = nt * 16 + nlo;
        float bb   = b2[ncol];
        v8f_t c;
#pragma unroll
        for (int r = 0; r < 8; ++r) c[r] = bb;
        c = wmma_f16(h0, wfrag(W2p, 0, 4, nt, lane), c);
        c = wmma_f16(h1, wfrag(W2p, 1, 4, nt, lane), c);
#pragma unroll
        for (int r = 0; r < 8; ++r)
            H[(mrow + r) * 64 + ncol] = (_Float16)softplus_f(c[r]);
    }
    lds_fence();

    // Layer 3: 64 -> 32, linear, f32 out
    v16h_t g0 = load_a16(H, 64, 0, lane);
    v16h_t g1 = load_a16(H, 64, 1, lane);
#pragma unroll
    for (int nt = 0; nt < 2; ++nt) {
        int   ncol = nt * 16 + nlo;
        float bb   = b3[ncol];
        v8f_t c;
#pragma unroll
        for (int r = 0; r < 8; ++r) c[r] = bb;
        c = wmma_f16(g0, wfrag(W3p, 0, 2, nt, lane), c);
        c = wmma_f16(g1, wfrag(W3p, 1, 2, nt, lane), c);
#pragma unroll
        for (int r = 0; r < 8; ++r)
            O[(mrow + r) * 32 + ncol] = c[r];
    }
    lds_fence();
}

// ---------------------------------------------------------------------------
// Scatter kernels (segment sums via f32 global atomics; count as float).
// ---------------------------------------------------------------------------
__global__ __launch_bounds__(256) void scatter_bond_kernel(
    const float* __restrict__ bf, const int* __restrict__ ba,
    float* __restrict__ sum, float* __restrict__ cnt, int Nb) {
    int gid = blockIdx.x * 256 + threadIdx.x;
    if (gid >= Nb * 32) return;
    int   bond = gid >> 5, f = gid & 31;
    float v  = bf[gid];
    long  a0 = ba[bond * 2 + 0];
    long  a1 = ba[bond * 2 + 1];
    atomicAdd(&sum[a0 * 32 + f], v);
    atomicAdd(&sum[a1 * 32 + f], v);
    if (f == 0) { atomicAdd(&cnt[a0], 1.f); atomicAdd(&cnt[a1], 1.f); }
}

__global__ __launch_bounds__(256) void scatter_seg_kernel(
    const float* __restrict__ vals, const int* __restrict__ seg,
    float* __restrict__ sum, float* __restrict__ cnt, int N) {
    int gid = blockIdx.x * 256 + threadIdx.x;
    if (gid >= N * 32) return;
    int  row = gid >> 5, f = gid & 31;
    long s = seg[row];
    atomicAdd(&sum[s * 32 + f], vals[gid]);
    if (f == 0) atomicAdd(&cnt[s], 1.f);
}

// ---------------------------------------------------------------------------
// MLP kernels: 128 threads = 4 waves, one 16-row tile per wave per loop step.
// Grid-stride tile loop amortizes the per-block weight packing.
// ---------------------------------------------------------------------------
__global__ __launch_bounds__(128) void atom_mlp_kernel(
    const float* __restrict__ atom_feats, const float* __restrict__ sumA,
    const float* __restrict__ cntA, const float* __restrict__ global_feats,
    const int* __restrict__ atom_mol,
    const float* __restrict__ W1, const float* __restrict__ b1,
    const float* __restrict__ W2, const float* __restrict__ b2,
    const float* __restrict__ W3, const float* __restrict__ b3,
    float* __restrict__ out, int Na) {
    __shared__ alignas(32) _Float16 W1p[3 * 4 * 512];
    __shared__ alignas(32) _Float16 W2p[2 * 4 * 512];
    __shared__ alignas(32) _Float16 W3p[2 * 2 * 512];
    __shared__ float bias1[64], bias2[64], bias3[32];
    __shared__ alignas(32) _Float16 Ain[4][16 * 96];
    __shared__ alignas(32) _Float16 Hst[4][16 * 64];
    __shared__ float Ost[4][16 * 32];

    int tid = threadIdx.x;
    pack_weights(W1, W1p, 96, 64, tid, 128);
    pack_weights(W2, W2p, 64, 64, tid, 128);
    pack_weights(W3, W3p, 64, 32, tid, 128);
    for (int i = tid; i < 64; i += 128) { bias1[i] = b1[i]; bias2[i] = b2[i]; }
    for (int i = tid; i < 32; i += 128) bias3[i] = b3[i];
    __syncthreads();

    int  wave = tid >> 5, lane = tid & 31;
    long tiles = ((long)Na + 15) / 16;
    for (long t = (long)blockIdx.x * 4 + wave; t < tiles;
         t += (long)gridDim.x * 4) {
        long base = t * 16;
        _Float16* A = Ain[wave];
        for (int i = lane; i < 512; i += 32) {
            int  r = i >> 5, c = i & 31;
            long row = base + r;
            float va = 0.f, vm = 0.f, vg = 0.f;
            if (row < Na) {
                va = atom_feats[row * 32 + c];
                float cn = cntA[row];
                vm = (cn > 0.f) ? sumA[row * 32 + c] * (1.f / cn) : 0.f;
                long mol = atom_mol[row];
                vg = global_feats[mol * 32 + c];
            }
            A[r * 96 + c]      = (_Float16)va;
            A[r * 96 + 32 + c] = (_Float16)vm;
            A[r * 96 + 64 + c] = (_Float16)vg;
        }
        lds_fence();
        mlp3<3>(A, W1p, bias1, W2p, bias2, W3p, bias3, Hst[wave], Ost[wave],
                lane);
        float* O = Ost[wave];
        for (int i = lane; i < 512; i += 32) {
            int  r = i >> 5, c = i & 31;
            long row = base + r;
            if (row < Na) out[row * 32 + c] = O[r * 32 + c];
        }
    }
}

__global__ __launch_bounds__(128) void bond_mlp_kernel(
    const float* __restrict__ bond_feats, const float* __restrict__ atom_new,
    const int* __restrict__ bond_atoms, const float* __restrict__ global_feats,
    const int* __restrict__ bond_mol,
    const float* __restrict__ W1, const float* __restrict__ b1,
    const float* __restrict__ W2, const float* __restrict__ b2,
    const float* __restrict__ W3, const float* __restrict__ b3,
    float* __restrict__ out, int Nb) {
    __shared__ alignas(32) _Float16 W1p[4 * 4 * 512];
    __shared__ alignas(32) _Float16 W2p[2 * 4 * 512];
    __shared__ alignas(32) _Float16 W3p[2 * 2 * 512];
    __shared__ float bias1[64], bias2[64], bias3[32];
    __shared__ alignas(32) _Float16 Ain[4][16 * 128];
    __shared__ alignas(32) _Float16 Hst[4][16 * 64];
    __shared__ float Ost[4][16 * 32];

    int tid = threadIdx.x;
    pack_weights(W1, W1p, 128, 64, tid, 128);
    pack_weights(W2, W2p, 64, 64, tid, 128);
    pack_weights(W3, W3p, 64, 32, tid, 128);
    for (int i = tid; i < 64; i += 128) { bias1[i] = b1[i]; bias2[i] = b2[i]; }
    for (int i = tid; i < 32; i += 128) bias3[i] = b3[i];
    __syncthreads();

    int  wave = tid >> 5, lane = tid & 31;
    long tiles = ((long)Nb + 15) / 16;
    for (long t = (long)blockIdx.x * 4 + wave; t < tiles;
         t += (long)gridDim.x * 4) {
        long base = t * 16;
        _Float16* A = Ain[wave];
        for (int i = lane; i < 512; i += 32) {
            int  r = i >> 5, c = i & 31;
            long row = base + r;
            float vb = 0.f, v0 = 0.f, v1 = 0.f, vg = 0.f;
            if (row < Nb) {
                vb = bond_feats[row * 32 + c];
                long a0 = bond_atoms[row * 2 + 0];
                long a1 = bond_atoms[row * 2 + 1];
                v0 = atom_new[a0 * 32 + c];
                v1 = atom_new[a1 * 32 + c];
                long mol = bond_mol[row];
                vg = global_feats[mol * 32 + c];
            }
            A[r * 128 + c]      = (_Float16)vb;
            A[r * 128 + 32 + c] = (_Float16)v0;
            A[r * 128 + 64 + c] = (_Float16)v1;
            A[r * 128 + 96 + c] = (_Float16)vg;
        }
        lds_fence();
        mlp3<4>(A, W1p, bias1, W2p, bias2, W3p, bias3, Hst[wave], Ost[wave],
                lane);
        float* O = Ost[wave];
        for (int i = lane; i < 512; i += 32) {
            int  r = i >> 5, c = i & 31;
            long row = base + r;
            if (row < Nb) out[row * 32 + c] = O[r * 32 + c];
        }
    }
}

__global__ __launch_bounds__(128) void global_mlp_kernel(
    const float* __restrict__ global_feats,
    const float* __restrict__ sumAg, const float* __restrict__ cntAg,
    const float* __restrict__ sumBg, const float* __restrict__ cntBg,
    const float* __restrict__ W1, const float* __restrict__ b1,
    const float* __restrict__ W2, const float* __restrict__ b2,
    const float* __restrict__ W3, const float* __restrict__ b3,
    float* __restrict__ out, int Ng) {
    __shared__ alignas(32) _Float16 W1p[3 * 4 * 512];
    __shared__ alignas(32) _Float16 W2p[2 * 4 * 512];
    __shared__ alignas(32) _Float16 W3p[2 * 2 * 512];
    __shared__ float bias1[64], bias2[64], bias3[32];
    __shared__ alignas(32) _Float16 Ain[4][16 * 96];
    __shared__ alignas(32) _Float16 Hst[4][16 * 64];
    __shared__ float Ost[4][16 * 32];

    int tid = threadIdx.x;
    pack_weights(W1, W1p, 96, 64, tid, 128);
    pack_weights(W2, W2p, 64, 64, tid, 128);
    pack_weights(W3, W3p, 64, 32, tid, 128);
    for (int i = tid; i < 64; i += 128) { bias1[i] = b1[i]; bias2[i] = b2[i]; }
    for (int i = tid; i < 32; i += 128) bias3[i] = b3[i];
    __syncthreads();

    int  wave = tid >> 5, lane = tid & 31;
    long tiles = ((long)Ng + 15) / 16;
    for (long t = (long)blockIdx.x * 4 + wave; t < tiles;
         t += (long)gridDim.x * 4) {
        long base = t * 16;
        _Float16* A = Ain[wave];
        for (int i = lane; i < 512; i += 32) {
            int  r = i >> 5, c = i & 31;
            long row = base + r;
            float vg = 0.f, va = 0.f, vb = 0.f;
            if (row < Ng) {
                vg = global_feats[row * 32 + c];
                float ca = cntAg[row];
                va = (ca > 0.f) ? sumAg[row * 32 + c] * (1.f / ca) : 0.f;
                float cb = cntBg[row];
                vb = (cb > 0.f) ? sumBg[row * 32 + c] * (1.f / cb) : 0.f;
            }
            A[r * 96 + c]      = (_Float16)vg;
            A[r * 96 + 32 + c] = (_Float16)va;
            A[r * 96 + 64 + c] = (_Float16)vb;
        }
        lds_fence();
        mlp3<3>(A, W1p, bias1, W2p, bias2, W3p, bias3, Hst[wave], Ost[wave],
                lane);
        float* O = Ost[wave];
        for (int i = lane; i < 512; i += 32) {
            int  r = i >> 5, c = i & 31;
            long row = base + r;
            if (row < Ng) out[row * 32 + c] = O[r * 32 + c];
        }
    }
}

// ---------------------------------------------------------------------------
extern "C" void kernel_launch(void* const* d_in, const int* in_sizes, int n_in,
                              void* d_out, int out_size, void* d_ws,
                              size_t ws_size, hipStream_t stream) {
    const float* atom_feats   = (const float*)d_in[0];
    const float* bond_feats   = (const float*)d_in[1];
    const float* global_feats = (const float*)d_in[2];
    const int*   bond_atoms   = (const int*)d_in[3];
    const int*   atom_mol     = (const int*)d_in[4];
    const int*   bond_mol     = (const int*)d_in[5];
    const float* aW1 = (const float*)d_in[6],  *ab1 = (const float*)d_in[7];
    const float* aW2 = (const float*)d_in[8],  *ab2 = (const float*)d_in[9];
    const float* aW3 = (const float*)d_in[10], *ab3 = (const float*)d_in[11];
    const float* bW1 = (const float*)d_in[12], *bb1 = (const float*)d_in[13];
    const float* bW2 = (const float*)d_in[14], *bb2 = (const float*)d_in[15];
    const float* bW3 = (const float*)d_in[16], *bb3 = (const float*)d_in[17];
    const float* gW1 = (const float*)d_in[18], *gb1 = (const float*)d_in[19];
    const float* gW2 = (const float*)d_in[20], *gb2 = (const float*)d_in[21];
    const float* gW3 = (const float*)d_in[22], *gb3 = (const float*)d_in[23];

    const int Na = in_sizes[0] / 32;
    const int Nb = in_sizes[1] / 32;
    const int Ng = in_sizes[2] / 32;

    // Workspace layout (f32): segment-sum accumulators + counts.
    float* ws = (float*)d_ws;
    size_t o = 0;
    float* sumA  = ws + o; o += (size_t)Na * 32;
    float* cntA  = ws + o; o += (size_t)Na;
    float* sumAg = ws + o; o += (size_t)Ng * 32;
    float* cntAg = ws + o; o += (size_t)Ng;
    float* sumBg = ws + o; o += (size_t)Ng * 32;
    float* cntBg = ws + o; o += (size_t)Ng;
    hipMemsetAsync(d_ws, 0, o * sizeof(float), stream);

    float* out_atom = (float*)d_out;
    float* out_bond = out_atom + (size_t)Na * 32;
    float* out_glob = out_bond + (size_t)Nb * 32;

    // 1) mean incident-bond features per atom
    scatter_bond_kernel<<<(Nb * 32 + 255) / 256, 256, 0, stream>>>(
        bond_feats, bond_atoms, sumA, cntA, Nb);

    // 2) atom MLP (WMMA)
    int blocksA = (int)(((long)Na + 15) / 16 + 3) / 4;
    if (blocksA > 4096) blocksA = 4096;
    atom_mlp_kernel<<<blocksA, 128, 0, stream>>>(
        atom_feats, sumA, cntA, global_feats, atom_mol,
        aW1, ab1, aW2, ab2, aW3, ab3, out_atom, Na);

    // 3) bond MLP (WMMA) — uses updated atoms
    int blocksB = (int)(((long)Nb + 15) / 16 + 3) / 4;
    if (blocksB > 4096) blocksB = 4096;
    bond_mlp_kernel<<<blocksB, 128, 0, stream>>>(
        bond_feats, out_atom, bond_atoms, global_feats, bond_mol,
        bW1, bb1, bW2, bb2, bW3, bb3, out_bond, Nb);

    // 4) per-molecule means of updated atoms & bonds
    scatter_seg_kernel<<<(Na * 32 + 255) / 256, 256, 0, stream>>>(
        out_atom, atom_mol, sumAg, cntAg, Na);
    scatter_seg_kernel<<<(Nb * 32 + 255) / 256, 256, 0, stream>>>(
        out_bond, bond_mol, sumBg, cntBg, Nb);

    // 5) global MLP (WMMA)
    int blocksG = (int)(((long)Ng + 15) / 16 + 3) / 4;
    if (blocksG > 1024) blocksG = 1024;
    global_mlp_kernel<<<blocksG, 128, 0, stream>>>(
        global_feats, sumAg, cntAg, sumBg, cntBg,
        gW1, gb1, gW2, gb2, gW3, gb3, out_glob, Ng);
}
